// MHSA_84954453115187
// MI455X (gfx1250) — compile-verified, matrix operand order checked by reading, not run on hardware
//
#include <hip/hip_runtime.h>
#include <hip/hip_bf16.h>

typedef __attribute__((ext_vector_type(16))) _Float16 v16h;
typedef __attribute__((ext_vector_type(8)))  _Float16 v8h;
typedef __attribute__((ext_vector_type(4)))  _Float16 v4h;
typedef __attribute__((ext_vector_type(8)))  float    v8f;
typedef __attribute__((ext_vector_type(4)))  float    v4f;

#define D_IN 1024
#define D_MODEL 1024
#define NUM_HEADS 16
#define DEPTH 64
#define BATCH 2
#define SEQ 2048
#define ROWS (BATCH*SEQ)   // 4096

// Load a 16x32 f16 WMMA fragment (A layout; also used for B via the symmetric
// column-per-lane layout). rowbase = pointer to this lane's row/column,
// k0 = K-chunk offset, hs = lane>>4 (half-wave select).
// Per ISA: lanes 0-15 hold K = {k0+0..7, k0+16..23}; lanes 16-31 hold
// K = {k0+8..15, k0+24..31}. Two 16-byte loads.
__device__ __forceinline__ v16h load_frag16(const _Float16* rowbase, int k0, int hs) {
  union { v16h v; v8h h[2]; } u;
  u.h[0] = *(const v8h*)(rowbase + k0 + 8*hs);
  u.h[1] = *(const v8h*)(rowbase + k0 + 16 + 8*hs);
  return u.v;
}

__device__ __forceinline__ v8f wmma_f16(v16h a, v16h b, v8f c) {
  return __builtin_amdgcn_wmma_f32_16x16x32_f16(false, a, false, b, (short)0, c,
                                                false, false);
}

// ---------------- conversion kernels ----------------

// x fp32 -> f16, vectorized (b128 load / b64 store)
__global__ void cvt_x_kernel(const float* __restrict__ x, _Float16* __restrict__ xh,
                             int n4) {
  int i = blockIdx.x * blockDim.x + threadIdx.x;
  const int stride = gridDim.x * blockDim.x;
  const v4f* x4 = (const v4f*)x;
  v4h* xh4 = (v4h*)xh;
  for (; i < n4; i += stride) {
    xh4[i] = __builtin_convertvector(x4[i], v4h);
  }
}

// wt[mat][out][in] = W[mat][in][out], via 32x32 LDS tile (coalesced both sides)
__global__ __launch_bounds__(256) void cvt_w_kernel(
    const float* __restrict__ Wq, const float* __restrict__ Wk,
    const float* __restrict__ Wv, _Float16* __restrict__ wt) {
  __shared__ float tile[32][33];
  const int mat = blockIdx.z;
  const float* W = (mat == 0) ? Wq : ((mat == 1) ? Wk : Wv);
  const int c0 = blockIdx.x * 32;   // in-dim tile origin
  const int o0 = blockIdx.y * 32;   // out-dim tile origin
  const int tx = threadIdx.x;       // 0..31
#pragma unroll
  for (int rr = threadIdx.y; rr < 32; rr += 8)
    tile[rr][tx] = W[(size_t)(c0 + rr) * D_MODEL + o0 + tx];   // coalesced in o
  __syncthreads();
  _Float16* wm = wt + (size_t)mat * D_MODEL * D_IN;
#pragma unroll
  for (int rr = threadIdx.y; rr < 32; rr += 8)
    wm[(size_t)(o0 + rr) * D_IN + c0 + tx] = (_Float16)tile[tx][rr]; // coalesced in c
}

// ---------------- QKV projection GEMM ----------------
// Each wave: 16x64 output tile. blockIdx.y selects matrix (q/k/v).
// Q,K written [b,h,n,d]; V written transposed [b,h,d,n].

__global__ __launch_bounds__(256) void qkv_gemm_kernel(
    const _Float16* __restrict__ xh, const _Float16* __restrict__ wt,
    const float* __restrict__ bq, const float* __restrict__ bk,
    const float* __restrict__ bv,
    _Float16* __restrict__ qh, _Float16* __restrict__ kh, _Float16* __restrict__ vth)
{
  const int mat  = blockIdx.y;
  const int lane = threadIdx.x;
  const int hs   = lane >> 4;
  const int lr   = lane & 15;
  const int g    = blockIdx.x * 8 + threadIdx.y;  // 0..4095 wave tiles
  const int mt   = g >> 4;                        // 256 row tiles
  const int ct   = g & 15;                        // 16 col tiles of 64
  const int row0 = mt * 16, col0 = ct * 64;

  const _Float16* wmat = wt + (size_t)mat * D_MODEL * D_IN;
  const float* bias = (mat == 0) ? bq : ((mat == 1) ? bk : bv);

  v8f acc[4] = {};
  const _Float16* arow = xh + (size_t)(row0 + lr) * D_IN;
  for (int k0 = 0; k0 < D_IN; k0 += 32) {
    v16h a = load_frag16(arow, k0, hs);
#pragma unroll
    for (int c = 0; c < 4; ++c) {
      v16h b = load_frag16(wmat + (size_t)(col0 + c * 16 + lr) * D_IN, k0, hs);
      acc[c] = wmma_f16(a, b, acc[c]);
    }
  }

#pragma unroll
  for (int c = 0; c < 4; ++c) {
    const int col = col0 + c * 16 + lr;
    const float bb = bias[col];
    const int h = col >> 6, d = col & 63;
#pragma unroll
    for (int r = 0; r < 8; ++r) {
      const int grow = row0 + r + 8 * hs;       // global token row
      const int b = grow >> 11, n = grow & (SEQ - 1);
      const _Float16 val = (_Float16)(acc[c][r] + bb);
      if (mat == 0)
        qh[((size_t)(b * NUM_HEADS + h) * SEQ + n) * DEPTH + d] = val;
      else if (mat == 1)
        kh[((size_t)(b * NUM_HEADS + h) * SEQ + n) * DEPTH + d] = val;
      else
        vth[((size_t)(b * NUM_HEADS + h) * DEPTH + d) * SEQ + n] = val;
    }
  }
}

// ---------------- flash attention ----------------
// One wave per 16-query tile per (b,h). 64-key inner step: 8 WMMAs for S,
// 8 WMMAs for O += P*V, one softmax reduction pass per step (exp2 domain).

__global__ __launch_bounds__(32) void attn_kernel(
    const _Float16* __restrict__ qh, const _Float16* __restrict__ kh,
    const _Float16* __restrict__ vth, float* __restrict__ out)
{
  __shared__ __align__(16) _Float16 pbuf[16 * 64];   // P tile 16x64
  const int lane = threadIdx.x;
  const int hs = lane >> 4;
  const int lr = lane & 15;

  const int blk = blockIdx.x;       // 0..4095
  const int mt  = blk & 127;        // 128 query tiles
  const int bh  = blk >> 7;         // 0..31
  const int h   = bh & (NUM_HEADS - 1);
  const int b   = bh >> 4;
  const int m0  = mt * 16;

  const _Float16* Qbh = qh  + (size_t)bh * SEQ * DEPTH;
  const _Float16* Kbh = kh  + (size_t)bh * SEQ * DEPTH;
  const _Float16* Vt  = vth + (size_t)bh * DEPTH * SEQ;  // [d][n]

  const v16h a0 = load_frag16(Qbh + (size_t)(m0 + lr) * DEPTH, 0, hs);
  const v16h a1 = load_frag16(Qbh + (size_t)(m0 + lr) * DEPTH, 32, hs);

  float rmax[8], rsum[8];
#pragma unroll
  for (int r = 0; r < 8; ++r) { rmax[r] = -1e30f; rsum[r] = 0.f; }
  v8f o[4] = {};

  // scores scaled into exp2 domain: 2^(s * 1/sqrt(64) * log2(e))
  const float scale2 = 0.125f * 1.44269504088896340736f;

  for (int j0 = 0; j0 < SEQ; j0 += 64) {
    // prefetch next key/value block (speculative; dropped if OOB)
    {
      const int jn = j0 + 64;
      __builtin_prefetch(Kbh + (size_t)(jn + lane) * DEPTH, 0, 1);
      __builtin_prefetch(Kbh + (size_t)(jn + 32 + lane) * DEPTH, 0, 1);
      __builtin_prefetch(Vt + (size_t)(lane) * SEQ + jn, 0, 1);
      __builtin_prefetch(Vt + (size_t)(32 + lane) * SEQ + jn, 0, 1);
    }

    // S = Q * K^T for 64 keys: 4 C tiles, depth 64 split into 2x K=32
    v8f s[4] = {v8f{}, v8f{}, v8f{}, v8f{}};
#pragma unroll
    for (int t = 0; t < 4; ++t) {
      const _Float16* krow = Kbh + (size_t)(j0 + t * 16 + lr) * DEPTH;
      s[t] = wmma_f16(a0, load_frag16(krow, 0, hs), s[t]);
      s[t] = wmma_f16(a1, load_frag16(krow, 32, hs), s[t]);
    }

    // online softmax (row r of this half-wave = global row r + 8*hs)
#pragma unroll
    for (int r = 0; r < 8; ++r) {
      float v0 = s[0][r] * scale2, v1 = s[1][r] * scale2;
      float v2 = s[2][r] * scale2, v3 = s[3][r] * scale2;
      float t = fmaxf(fmaxf(v0, v1), fmaxf(v2, v3));
      for (int mm = 1; mm < 16; mm <<= 1) t = fmaxf(t, __shfl_xor(t, mm, 32));
      const float nm   = fmaxf(rmax[r], t);
      const float corr = exp2f(rmax[r] - nm);
      const float p0 = exp2f(v0 - nm);
      const float p1 = exp2f(v1 - nm);
      const float p2 = exp2f(v2 - nm);
      const float p3 = exp2f(v3 - nm);
      rmax[r] = nm;
      float ps = (p0 + p1) + (p2 + p3);
      for (int mm = 1; mm < 16; mm <<= 1) ps += __shfl_xor(ps, mm, 32);
      rsum[r] = rsum[r] * corr + ps;
#pragma unroll
      for (int c = 0; c < 4; ++c) o[c][r] *= corr;
      const int m = r + 8 * hs;
      pbuf[m * 64 + lr]      = (_Float16)p0;   // C layout -> [m][k] in LDS
      pbuf[m * 64 + 16 + lr] = (_Float16)p1;
      pbuf[m * 64 + 32 + lr] = (_Float16)p2;
      pbuf[m * 64 + 48 + lr] = (_Float16)p3;
    }
    __syncthreads();

    // P (16x64) as two A fragments from LDS; O += P * V (4 depth tiles of 16)
    const v16h pa0 = load_frag16(pbuf + lr * 64, 0, hs);
    const v16h pa1 = load_frag16(pbuf + lr * 64, 32, hs);
#pragma unroll
    for (int c = 0; c < 4; ++c) {
      const _Float16* vrow = Vt + (size_t)(c * 16 + lr) * SEQ;
      o[c] = wmma_f16(pa0, load_frag16(vrow, j0, hs), o[c]);
      o[c] = wmma_f16(pa1, load_frag16(vrow, j0 + 32, hs), o[c]);
    }
    __syncthreads();
  }

  // normalize and store fp32 output [b][n][h*64+d]
#pragma unroll
  for (int r = 0; r < 8; ++r) {
    const float inv = 1.f / rsum[r];
    const int n = m0 + r + 8 * hs;
#pragma unroll
    for (int c = 0; c < 4; ++c) {
      out[((size_t)(b * SEQ + n)) * D_MODEL + h * DEPTH + c * 16 + lr] =
          o[c][r] * inv;
    }
  }
}

// ---------------- launcher ----------------

extern "C" void kernel_launch(void* const* d_in, const int* in_sizes, int n_in,
                              void* d_out, int out_size, void* d_ws, size_t ws_size,
                              hipStream_t stream) {
  (void)in_sizes; (void)n_in; (void)out_size; (void)ws_size;
  const float* x  = (const float*)d_in[0];
  const float* Wq = (const float*)d_in[1];
  const float* bq = (const float*)d_in[2];
  const float* Wk = (const float*)d_in[3];
  const float* bk = (const float*)d_in[4];
  const float* Wv = (const float*)d_in[5];
  const float* bv = (const float*)d_in[6];
  float* out = (float*)d_out;

  char* ws = (char*)d_ws;
  _Float16* xh  = (_Float16*)(ws);                      //  8 MiB: x in f16
  _Float16* wt  = (_Float16*)(ws + ((size_t)8  << 20)); //  6 MiB: W^T f16 x3
  _Float16* qh  = (_Float16*)(ws + ((size_t)14 << 20)); //  8 MiB: Q [b,h,n,d]
  _Float16* kh  = (_Float16*)(ws + ((size_t)22 << 20)); //  8 MiB: K [b,h,n,d]
  _Float16* vth = (_Float16*)(ws + ((size_t)30 << 20)); //  8 MiB: V [b,h,d,n]

  cvt_x_kernel<<<1024, 256, 0, stream>>>(x, xh, ROWS * D_IN / 4);

  dim3 wgrid(32, 32, 3), wblk(32, 8);
  cvt_w_kernel<<<wgrid, wblk, 0, stream>>>(Wq, Wk, Wv, wt);

  dim3 gb(512, 3), tb(32, 8);
  qkv_gemm_kernel<<<gb, tb, 0, stream>>>(xh, wt, bq, bk, bv, qh, kh, vth);

  attn_kernel<<<4096, 32, 0, stream>>>(qh, kh, vth, out);
}
